// BahdanauAttention_34651796144654
// MI455X (gfx1250) — matrix operand order repaired and checked by reading, not compiled, not run on hardware
//
#include <hip/hip_runtime.h>
#include <hip/hip_bf16.h>
#include <stdint.h>

// ---------------------------------------------------------------------------
// Bahdanau attention, fused for MI455X (gfx1250, wave32, WMMA bf16 + TDM)
//   enc:  [64, 2048, 512] f32   dh: [64,512]   W1,W2: [512,512]  b1,b2,V: [512]
//   out = context [64,512] ++ attn [64,2048]   (f32)
// ---------------------------------------------------------------------------

typedef __attribute__((ext_vector_type(16))) __bf16 v16bf;
typedef __attribute__((ext_vector_type(8)))  float  v8f;
typedef __attribute__((ext_vector_type(4)))  unsigned int v4u;
typedef __attribute__((ext_vector_type(8)))  int    v8i;
typedef __attribute__((ext_vector_type(4)))  int    v4i;

#define HDIM 512
#define SEQ  2048
#define BATCH 64

// ---------------- kernel 0a: W1 (f32) -> bf16 ------------------------------
__global__ void k_cvt_w1(const float* __restrict__ W1, __bf16* __restrict__ W1bf) {
    int i = blockIdx.x * 256 + threadIdx.x;         // grid covers 512*512
    W1bf[i] = (__bf16)W1[i];
}

// ---------------- kernel 0b: dec_proj = dh @ W2^T + b2 ---------------------
__global__ void k_dproj(const float* __restrict__ dh, const float* __restrict__ W2,
                        const float* __restrict__ b2, float* __restrict__ dproj) {
    int idx = blockIdx.x * 256 + threadIdx.x;       // 64*512 outputs
    int b = idx >> 9;
    int o = idx & 511;
    const float* h = dh + b * HDIM;
    const float* w = W2 + (size_t)o * HDIM;
    float acc = b2[o];
    #pragma unroll 4
    for (int k = 0; k < HDIM; ++k) acc += h[k] * w[k];
    dproj[idx] = acc;
}

// ---- TDM: load one 16-row x 512-col bf16 tile of W1 into LDS --------------
// D# per CDNA5 ISA ch.8: group0 = {count, lds_addr, global_addr, type=2},
// group1 = {data_size=2B, tensor_dim0/1=512, tile_dim0=512, tile_dim1=16,
//           tensor_dim0_stride=512}, groups 2/3 = 0 (2-D tensor).
// This toolchain (clang-23 / therock-10.0) exposes the 6-arg builtin form.
static __device__ __forceinline__ void tdm_load_b_tile(const __bf16* gptr,
                                                       unsigned int lds_off) {
    uint64_t ga = (uint64_t)(uintptr_t)gptr;
    v4u g0;
    g0[0] = 1u;                                         // count=1, user descriptor
    g0[1] = lds_off;                                    // lds_addr (bytes)
    g0[2] = (unsigned int)(ga & 0xffffffffu);           // global_addr[31:0]
    g0[3] = (unsigned int)((ga >> 32) & 0x01ffffffu)    // global_addr[56:32]
          | (2u << 30);                                 // type = 2 ("image")
    v8i g1;
    g1[0] = (int)(1u << 16);          // data_size = 1 (2 bytes/elem)
    g1[1] = (int)(512u << 16);        // tensor_dim0 = 512 (elems, contiguous)
    g1[2] = (int)(512u << 16);        // tensor_dim1 = 512 rows
    g1[3] = (int)(512u << 16);        // tile_dim0 = 512 (full row)
    g1[4] = 16;                       // tile_dim1 = 16 rows, tile_dim2 = 0
    g1[5] = 512;                      // tensor_dim0_stride = 512 elems
    g1[6] = 0;                        // stride0 hi / tensor_dim1_stride lo
    g1[7] = 0;                        // tensor_dim1_stride hi
    v4i z4 = {0, 0, 0, 0};
    v8i z8 = {0, 0, 0, 0, 0, 0, 0, 0};
    __builtin_amdgcn_tensor_load_to_lds(g0, g1, z4, z4, z8, 0);
}

// ---------------- kernel 1: scores via bf16 WMMA + TDM-staged B ------------
// One wave owns 16 encoder rows; block = 4 waves = 64 rows of one batch.
// A-panel (16 x 512) in registers (16 v16bf frags). B tiles (16 N-cols x
// 512 K) are DMA'd into LDS by wave 0 (double-buffered, tensorcnt-synced);
// all 4 waves consume the same tile -> 4x less L2 B-traffic.
__global__ __launch_bounds__(128) void k_scores(
        const float*  __restrict__ enc,
        const __bf16* __restrict__ W1bf,
        const float*  __restrict__ b1,
        const float*  __restrict__ Vw,
        const float*  __restrict__ dproj,
        float*        __restrict__ scores) {
    __shared__ __align__(128) __bf16 bbuf[2][16 * HDIM];   // 2 x 16 KB

    const int b    = blockIdx.y;
    const int wave = threadIdx.x >> 5;
    const int lane = threadIdx.x & 31;
    const int half = lane >> 4;          // which 16-lane half
    const int lr   = lane & 15;
    const int s0   = blockIdx.x * 64 + wave * 16;

    // --- kick off DMA of B tile 0 while we build the A panel ---------------
    if (wave == 0)
        tdm_load_b_tile(W1bf, (unsigned int)(uintptr_t)&bbuf[0][0]);

    // --- load A panel: row (s0+lr), bf16-convert into WMMA A layout --------
    // lanes 0-15: K in {0..15}; lanes 16-31: K in {16..31}
    const float* arow  = enc + ((size_t)b * SEQ + s0 + lr) * HDIM;
    const int    koff1 = half ? 16 : 0;
    const int    koff2 = half ? 24 : 8;

    v16bf afrag[16];
    #pragma unroll
    for (int kt = 0; kt < 16; ++kt) {
        const float* p = arow + kt * 32;
        float4 a0 = *(const float4*)(p + koff1);
        float4 a1 = *(const float4*)(p + koff1 + 4);
        float4 a2 = *(const float4*)(p + koff2);
        float4 a3 = *(const float4*)(p + koff2 + 4);
        v16bf f;
        f[0]  = (__bf16)a0.x; f[1]  = (__bf16)a0.y; f[2]  = (__bf16)a0.z; f[3]  = (__bf16)a0.w;
        f[4]  = (__bf16)a1.x; f[5]  = (__bf16)a1.y; f[6]  = (__bf16)a1.z; f[7]  = (__bf16)a1.w;
        f[8]  = (__bf16)a2.x; f[9]  = (__bf16)a2.y; f[10] = (__bf16)a2.z; f[11] = (__bf16)a2.w;
        f[12] = (__bf16)a3.x; f[13] = (__bf16)a3.y; f[14] = (__bf16)a3.z; f[15] = (__bf16)a3.w;
        afrag[kt] = f;
    }

    float partial[8];
    #pragma unroll
    for (int r = 0; r < 8; ++r) partial[r] = 0.0f;

    #pragma unroll 1
    for (int nt = 0; nt < 32; ++nt) {
        if (wave == 0)
            __builtin_amdgcn_s_wait_tensorcnt(0);   // tile nt landed in LDS
        __syncthreads();
        if (wave == 0 && nt + 1 < 32)               // prefetch tile nt+1
            tdm_load_b_tile(W1bf + (size_t)(nt + 1) * 16 * HDIM,
                            (unsigned int)(uintptr_t)&bbuf[(nt + 1) & 1][0]);

        // B fragment: column n = nt*16+lr, K-contiguous 32B per lane from LDS
        const __bf16* brow = &bbuf[nt & 1][0] + lr * HDIM + half * 16;
        v8f acc = {};
        #pragma unroll
        for (int kt = 0; kt < 16; ++kt) {
            v16bf bf = *(const v16bf*)(brow + kt * 32);
            acc = __builtin_amdgcn_wmma_f32_16x16x32_bf16(
                      false, afrag[kt], false, bf, (short)0, acc, false, false);
        }
        const int   n   = nt * 16 + lr;
        const float add = b1[n] + dproj[b * HDIM + n];
        const float vn  = Vw[n];
        // acc[r] = energy(M = half*8 + r, N = n)
        #pragma unroll
        for (int r = 0; r < 8; ++r)
            partial[r] += tanhf(acc[r] + add) * vn;
        __syncthreads();                            // done reading bbuf[nt&1]
    }

    // reduce over the 16 lanes of each half (xor masks < 16 stay within a half)
    #pragma unroll
    for (int r = 0; r < 8; ++r) {
        float v = partial[r];
        v += __shfl_xor(v, 1, 32);
        v += __shfl_xor(v, 2, 32);
        v += __shfl_xor(v, 4, 32);
        v += __shfl_xor(v, 8, 32);
        partial[r] = v;
    }
    if (lr == 0) {
        #pragma unroll
        for (int r = 0; r < 8; ++r)
            scores[(size_t)b * SEQ + s0 + half * 8 + r] = partial[r];
    }
}

// ---------------- kernel 2: softmax over S + context -----------------------
__global__ __launch_bounds__(256) void k_softmax_ctx(
        const float* __restrict__ enc, const float* __restrict__ scores,
        float* __restrict__ out_ctx, float* __restrict__ out_attn) {
    __shared__ float sm[SEQ];
    __shared__ float red[8];
    const int b = blockIdx.x;
    const int t = threadIdx.x;

    float m = -3.4e38f;
    for (int i = t; i < SEQ; i += 256) {
        float v = scores[(size_t)b * SEQ + i];
        sm[i] = v;
        m = fmaxf(m, v);
    }
    for (int o = 16; o >= 1; o >>= 1) m = fmaxf(m, __shfl_xor(m, o, 32));
    if ((t & 31) == 0) red[t >> 5] = m;
    __syncthreads();
    float bm = red[0];
    #pragma unroll
    for (int i = 1; i < 8; ++i) bm = fmaxf(bm, red[i]);
    __syncthreads();

    float s = 0.0f;
    for (int i = t; i < SEQ; i += 256) {
        float e = __expf(sm[i] - bm);
        sm[i] = e;
        s += e;
    }
    for (int o = 16; o >= 1; o >>= 1) s += __shfl_xor(s, o, 32);
    if ((t & 31) == 0) red[t >> 5] = s;
    __syncthreads();
    float tot = 0.0f;
    #pragma unroll
    for (int i = 0; i < 8; ++i) tot += red[i];
    const float inv = 1.0f / tot;

    for (int i = t; i < SEQ; i += 256) {
        float a = sm[i] * inv;
        sm[i] = a;
        out_attn[(size_t)b * SEQ + i] = a;
    }
    __syncthreads();

    // context[b][h] = sum_s attn[s] * enc[b][s][h]; thread t owns h = t, t+256
    float c0 = 0.0f, c1 = 0.0f;
    const float* eb = enc + (size_t)b * SEQ * HDIM;
    for (int si = 0; si < SEQ; ++si) {
        const float a = sm[si];
        c0 += a * eb[(size_t)si * HDIM + t];
        c1 += a * eb[(size_t)si * HDIM + t + 256];
    }
    out_ctx[b * HDIM + t]       = c0;
    out_ctx[b * HDIM + t + 256] = c1;
}

// ---------------------------------------------------------------------------
extern "C" void kernel_launch(void* const* d_in, const int* in_sizes, int n_in,
                              void* d_out, int out_size, void* d_ws, size_t ws_size,
                              hipStream_t stream) {
    const float* enc = (const float*)d_in[0];
    const float* dh  = (const float*)d_in[1];
    const float* W1  = (const float*)d_in[2];
    const float* b1  = (const float*)d_in[3];
    const float* W2  = (const float*)d_in[4];
    const float* b2  = (const float*)d_in[5];
    const float* Vw  = (const float*)d_in[6];
    // d_in[7] (bv): additive constant to all scores -> softmax-invariant, unused.

    char* ws = (char*)d_ws;
    __bf16* W1bf  = (__bf16*)ws;                                   // 512 KB
    float*  dproj = (float*)(ws + 512 * 1024);                     // 128 KB
    float*  scr   = (float*)(ws + 512 * 1024 + 128 * 1024);        // 512 KB

    float* out_ctx  = (float*)d_out;                // [64, 512]
    float* out_attn = out_ctx + BATCH * HDIM;       // [64, 2048]

    k_cvt_w1<<<(HDIM * HDIM) / 256, 256, 0, stream>>>(W1, W1bf);
    k_dproj<<<(BATCH * HDIM) / 256, 256, 0, stream>>>(dh, W2, b2, dproj);
    k_scores<<<dim3(SEQ / 64, BATCH), 128, 0, stream>>>(enc, W1bf, b1, Vw, dproj, scr);
    k_softmax_ctx<<<BATCH, 256, 0, stream>>>(enc, scr, out_ctx, out_attn);
}
